// SpanQualifier_19370302505488
// MI455X (gfx1250) — matrix operand order repaired and checked by reading, not compiled, not run on hardware
//
#include <hip/hip_runtime.h>

// ---------------------------------------------------------------------------
// SpanQualifier forward for MI455X (gfx1250, wave32, bf16 WMMA f32-accum,
// async global->LDS weight staging).
// B=4, L=192, D1=768, D2=256. Output S: [B,L,L] f32 (147456 elems).
//
// Input flattening (jax tree, sorted dict keys at every level):
//  0 H_c 1 H_cls 2 masks | 3-6 be_e(b1,b2,w1,w2) | 7-10 be_s | 11 conv_b
//  12 conv_w | 13-16 se_e(be,g,sb,sw) | 17-20 se_s | 21-24 se_x
//  25-28 ss_e.cls(unused) 29-32 ss_e.score(b1,b2,w1,w2)
//  33-36 ss_s.cls(unused) 37-40 ss_s.score
//  41-44 ss_x.cls(unused) 45-48 ss_x.score
//  49-50 w2e(b,w) 51-52 w2s(b,w) | 53 pos_e 54 pos_s 55 pos_x 56 tri
// ---------------------------------------------------------------------------

typedef __bf16 bf16;
typedef __bf16 bf16x16 __attribute__((ext_vector_type(16)));
typedef __bf16 bf16x8  __attribute__((ext_vector_type(8)));
typedef __bf16 bf16x4  __attribute__((ext_vector_type(4)));
typedef float  floatx8 __attribute__((ext_vector_type(8)));

union FragU { bf16x16 v; bf16x8 h[2]; };

__device__ __forceinline__ bf16 tobf(float x) { return (bf16)x; }

#define LL 192
#define D2C 256

// Flat-LDS addresses truncate to the wave-relative LDS byte offset (ISA
// aperture mapping), so the low 32 bits of a generic shared pointer are the
// LDS address the async-load VDST operand expects.
__device__ __forceinline__ unsigned ldsoff(const void* p) {
  return (unsigned)(unsigned long long)p;
}

// Per-lane async copy of 16 bytes global -> LDS (tracked by ASYNCcnt).
__device__ __forceinline__ void async_copy16(unsigned lds, const void* gaddr) {
  asm volatile("global_load_async_to_lds_b128 %0, %1, off"
               :: "v"(lds), "v"(gaddr) : "memory");
}

__device__ __forceinline__ void wait_async_le8() {
  asm volatile("s_wait_asynccnt 0x8" ::: "memory");
}
__device__ __forceinline__ void wait_async_0() {
  asm volatile("s_wait_asynccnt 0x0" ::: "memory");
}

// ---------------------------------------------------------------------------
// Pre-convert 256x256 f32 weight blocks to transposed bf16:
// dst[n*256+k] = bf16(src[k*256+n]). grid=(256,batch), block=256.
// ---------------------------------------------------------------------------
__global__ void __launch_bounds__(256)
w_to_bf16_t(const float* __restrict__ src, bf16* __restrict__ dst)
{
  long base = (long)blockIdx.y * 65536;
  int idx = blockIdx.x * 256 + threadIdx.x;   // 0..65535
  int k = idx >> 8, n = idx & 255;
  dst[base + (long)n * 256 + k] = tobf(src[base + idx]);
}

// ---------------------------------------------------------------------------
// Generic tiled GEMM: C[M,N] = act(A[M,K] @ W[K,N] + bias), bf16 WMMA f32 acc.
// 128 threads = 4 waves, tile 64x64, K step 32. M%64==N%64==0, K%32==0.
// Batched via blockIdx.z with element strides.
// ---------------------------------------------------------------------------
__global__ void __launch_bounds__(128)
gemm_bias_act(const float* __restrict__ A, const float* __restrict__ W,
              const float* __restrict__ bias, float* __restrict__ C,
              int M, int N, int K, long sA, long sW, long sC, int act)
{
  __shared__ __align__(16) bf16 Abuf[64][32];   // [m][k]
  __shared__ __align__(16) bf16 BTbuf[64][32];  // [n][k]
  (void)M;
  int b = blockIdx.z;
  A += (long)b * sA; W += (long)b * sW; C += (long)b * sC;
  int n0 = blockIdx.x * 64;
  int m0 = blockIdx.y * 64;
  int tid = threadIdx.x;
  int wave = tid >> 5, lane = tid & 31, lr = lane & 15, lg = lane >> 4;

  floatx8 acc[4] = {};
  for (int k0 = 0; k0 < K; k0 += 32) {
#pragma unroll
    for (int e = 0; e < 4; ++e) {              // A tile: float4 loads, b64 stores
      int idx = tid + 128 * e;                 // 0..511 quads
      int r = idx >> 3, c4 = (idx & 7) * 4;
      float4 v = *(const float4*)(A + (long)(m0 + r) * K + k0 + c4);
      bf16x4 o = { tobf(v.x), tobf(v.y), tobf(v.z), tobf(v.w) };
      *(bf16x4*)&Abuf[r][c4] = o;
    }
#pragma unroll
    for (int e = 0; e < 4; ++e) {              // W tile -> [n][k], b64 stores
      int idx = tid + 128 * e;
      int n = idx & 63, kq = (idx >> 6) * 4;   // kq in {0..28}
      const float* wp = W + (long)(k0 + kq) * N + n0 + n;
      bf16x4 o = { tobf(wp[0]), tobf(wp[(long)N]),
                   tobf(wp[2L * N]), tobf(wp[3L * N]) };
      *(bf16x4*)&BTbuf[n][kq] = o;
    }
    __syncthreads();
    FragU fa;
    const bf16* ap = &Abuf[wave * 16 + lr][0];
    fa.h[0] = *(const bf16x8*)(ap + 16 * lg);
    fa.h[1] = *(const bf16x8*)(ap + 16 * lg + 8);
#pragma unroll
    for (int t = 0; t < 4; ++t) {
      FragU fb;
      const bf16* bp = &BTbuf[t * 16 + lr][16 * lg];
      fb.h[0] = *(const bf16x8*)(bp);
      fb.h[1] = *(const bf16x8*)(bp + 8);
      acc[t] = __builtin_amdgcn_wmma_f32_16x16x32_bf16(
          false, fa.v, false, fb.v, (short)0, acc[t], false, false);
    }
    __syncthreads();
  }
#pragma unroll
  for (int t = 0; t < 4; ++t) {
    int col = n0 + t * 16 + lr;
    float bv = bias ? bias[col] : 0.f;
#pragma unroll
    for (int v = 0; v < 8; ++v) {
      int row = m0 + wave * 16 + v + 8 * lg;
      float x = acc[t][v] + bv;
      if (act) x = fmaxf(x, 0.f);
      C[(long)row * N + col] = x;
    }
  }
}

// ---------------------------------------------------------------------------
// Fused score MLP: spart[p] = sum_c relu(X @ W1 + b1)[p,c] * w2[c].
// X: LDS [64][256] bf16. W1T: global bf16, pre-transposed [n][k].
// Stages one 64x256 W1T block per n-pass via ASYNC global->LDS copies;
// w2-dot folded into the epilogue via ds_add_f32.
// ---------------------------------------------------------------------------
__device__ __forceinline__ void score_mlp64(
    const bf16 (*X)[D2C], bf16 (*Wst)[D2C], float* spart,
    const bf16* __restrict__ W1T, const float* __restrict__ b1,
    const float* __restrict__ w2)
{
  int tid = threadIdx.x;
  int wave = tid >> 5, lane = tid & 31, lr = lane & 15, lg = lane >> 4;
  if (tid < 64) spart[tid] = 0.f;
  __syncthreads();
  for (int nb = 0; nb < 4; ++nb) {
#pragma unroll
    for (int e = 0; e < 16; ++e) {            // 2048 async b128 copies / 128 thr
      int idx = tid + 128 * e;
      int row = idx >> 5, ch = (idx & 31) * 8;
      async_copy16(ldsoff(&Wst[row][ch]),
                   W1T + (long)(nb * 64 + row) * 256 + ch);
    }
    wait_async_0();
    __syncthreads();
    floatx8 acc[4] = {};
    for (int k0 = 0; k0 < 256; k0 += 32) {
      FragU fa;
      const bf16* ap = &X[wave * 16 + lr][k0];
      fa.h[0] = *(const bf16x8*)(ap + 16 * lg);
      fa.h[1] = *(const bf16x8*)(ap + 16 * lg + 8);
#pragma unroll
      for (int t = 0; t < 4; ++t) {
        FragU fb;
        const bf16* bp = &Wst[t * 16 + lr][k0 + 16 * lg];
        fb.h[0] = *(const bf16x8*)(bp);
        fb.h[1] = *(const bf16x8*)(bp + 8);
        acc[t] = __builtin_amdgcn_wmma_f32_16x16x32_bf16(
            false, fa.v, false, fb.v, (short)0, acc[t], false, false);
      }
    }
#pragma unroll
    for (int v = 0; v < 8; ++v) {
      float part = 0.f;
#pragma unroll
      for (int t = 0; t < 4; ++t) {
        int col = nb * 64 + t * 16 + lr;
        float h = fmaxf(acc[t][v] + b1[col], 0.f);
        part += h * w2[col];
      }
      atomicAdd(&spart[wave * 16 + v + 8 * lg], part);
    }
    __syncthreads();
  }
}

// ---------------------------------------------------------------------------
// Span-enum + score (paths s and e). Block: (b, i, 64-wide j tile).
// ---------------------------------------------------------------------------
__global__ void __launch_bounds__(128)
span_score_kernel(const float* __restrict__ Pi, const float* __restrict__ Pj,
                  const float* __restrict__ pos,
                  const float* __restrict__ g, const float* __restrict__ beta,
                  const bf16* __restrict__ W1T, const float* __restrict__ b1,
                  const float* __restrict__ w2, const float* __restrict__ b2,
                  const float* __restrict__ masks, const float* __restrict__ tri,
                  float* __restrict__ G, int transpose_out)
{
  __shared__ __align__(16) bf16 X[64][D2C];
  __shared__ __align__(16) bf16 stage[64 * D2C];
  __shared__ float spart[64];
  __shared__ __align__(16) float birow[D2C];
  int j0 = blockIdx.x * 64;
  int i  = blockIdx.y;
  int b  = blockIdx.z;
  int tid = threadIdx.x;
  for (int c = tid; c < D2C; c += 128)
    birow[c] = Pi[((long)(b * LL + i)) * D2C + c];
  __syncthreads();
  if (tid < 64) {
    int j = j0 + tid;
    const float4* pj4 = (const float4*)&Pj[((long)(b * LL + j)) * D2C];
    const float4* pp4 = (const float4*)&pos[((long)i * LL + j) * D2C];
    const float4* bi4 = (const float4*)birow;
    const float4* g4  = (const float4*)g;
    const float4* be4 = (const float4*)beta;
    float s = 0.f, ss = 0.f;
    for (int c = 0; c < 64; ++c) {
      float4 a = bi4[c], d = pj4[c], p = pp4[c];
      float x0 = a.x + d.x + p.x, x1 = a.y + d.y + p.y;
      float x2 = a.z + d.z + p.z, x3 = a.w + d.w + p.w;
      s  += x0 + x1 + x2 + x3;
      ss += x0 * x0 + x1 * x1 + x2 * x2 + x3 * x3;
    }
    float mu = s * (1.f / 256.f);
    float var = ss * (1.f / 256.f) - mu * mu;
    float rstd = rsqrtf(var + 1e-12f);
    for (int c = 0; c < 64; ++c) {
      float4 a = bi4[c], d = pj4[c], p = pp4[c], gg = g4[c], bb = be4[c];
      bf16x4 o = { tobf((a.x + d.x + p.x - mu) * rstd * gg.x + bb.x),
                   tobf((a.y + d.y + p.y - mu) * rstd * gg.y + bb.y),
                   tobf((a.z + d.z + p.z - mu) * rstd * gg.z + bb.z),
                   tobf((a.w + d.w + p.w - mu) * rstd * gg.w + bb.w) };
      *(bf16x4*)&X[tid][c * 4] = o;
    }
  }
  __syncthreads();
  score_mlp64(X, (bf16(*)[D2C])stage, spart, W1T, b1, w2);
  if (tid < 64) {
    int j = j0 + tid;
    float S = spart[tid] + b2[0];
    float mm = masks[b * LL + i] * masks[b * LL + j] * tri[i * LL + j];
    float val = S - 10000.f * (1.f - mm);
    if (transpose_out) G[((long)(b * LL + j)) * LL + i] = val;
    else               G[((long)(b * LL + i)) * LL + j] = val;
  }
}

// ---------------------------------------------------------------------------
// Row softmax over 192 columns; one row per block (grid = B*L).
// ---------------------------------------------------------------------------
__global__ void __launch_bounds__(256)
softmax192(float* __restrict__ G)
{
  __shared__ float red[256];
  int t = threadIdx.x;
  long row = blockIdx.x;
  float v = (t < LL) ? G[row * LL + t] : -3.4e38f;
  red[t] = v;
  __syncthreads();
  for (int s = 128; s >= 1; s >>= 1) {
    if (t < s) red[t] = fmaxf(red[t], red[t + s]);
    __syncthreads();
  }
  float m = red[0];
  __syncthreads();
  float e = (t < LL) ? __expf(v - m) : 0.f;
  red[t] = e;
  __syncthreads();
  for (int s = 128; s >= 1; s >>= 1) {
    if (t < s) red[t] += red[t + s];
    __syncthreads();
  }
  float inv = 1.f / red[0];
  if (t < LL) G[row * LL + t] = e * inv;
}

// ---------------------------------------------------------------------------
// Fused span-enum + 5x5 conv (implicit GEMM) + score MLP + mask.
// Block: 8x8 output tile; 12x12x256 bf16 LN'd patch in LDS. Conv keeps all
// 16 accumulators (full N=256). Weight slabs [256co][32ci] are double-buffered
// and prefetched with ASYNC global->LDS copies pipelined against the WMMAs.
// ---------------------------------------------------------------------------
__global__ void __launch_bounds__(128)
conv_score_kernel(const float* __restrict__ Bs2x, const float* __restrict__ Be2x,
                  const float* __restrict__ posx,
                  const float* __restrict__ g, const float* __restrict__ beta,
                  const bf16* __restrict__ convwT,  // [25][co 256][ci 256] bf16
                  const float* __restrict__ convb,
                  const bf16* __restrict__ W1T, const float* __restrict__ b1,
                  const float* __restrict__ w2, const float* __restrict__ b2,
                  const float* __restrict__ masks, const float* __restrict__ tri,
                  float* __restrict__ Sout)
{
  __shared__ __align__(16) bf16 patch[144][D2C];   // 72 KB
  __shared__ __align__(16) bf16 Y[64][D2C];        // 32 KB
  __shared__ __align__(16) bf16 stage[64 * D2C];   // 32 KB (2x16KB conv / mlp)
  __shared__ float spart[64];
  int j0 = blockIdx.x * 8, i0 = blockIdx.y * 8, b = blockIdx.z;
  int tid = threadIdx.x;
  int wave = tid >> 5, lane = tid & 31, lr = lane & 15, lg = lane >> 4;

  // 1. Recompute LayerNorm'd Mx patch with halo (zero pad at grid edges).
  for (int rr = tid; rr < 144; rr += 128) {
    int pi = rr / 12, pj = rr % 12;
    int ii = i0 - 2 + pi, jj = j0 - 2 + pj;
    if ((unsigned)ii >= (unsigned)LL || (unsigned)jj >= (unsigned)LL) {
      for (int c = 0; c < 64; ++c) {
        bf16x4 z = { tobf(0.f), tobf(0.f), tobf(0.f), tobf(0.f) };
        *(bf16x4*)&patch[rr][c * 4] = z;
      }
    } else {
      const float4* ps4 = (const float4*)&Bs2x[((long)(b * LL + ii)) * D2C];
      const float4* pe4 = (const float4*)&Be2x[((long)(b * LL + jj)) * D2C];
      const float4* pp4 = (const float4*)&posx[((long)ii * LL + jj) * D2C];
      const float4* g4  = (const float4*)g;
      const float4* be4 = (const float4*)beta;
      float s = 0.f, ss = 0.f;
      for (int c = 0; c < 64; ++c) {
        float4 a = ps4[c], d = pe4[c], p = pp4[c];
        float x0 = a.x + d.x + p.x, x1 = a.y + d.y + p.y;
        float x2 = a.z + d.z + p.z, x3 = a.w + d.w + p.w;
        s  += x0 + x1 + x2 + x3;
        ss += x0 * x0 + x1 * x1 + x2 * x2 + x3 * x3;
      }
      float mu = s * (1.f / 256.f);
      float var = ss * (1.f / 256.f) - mu * mu;
      float rstd = rsqrtf(var + 1e-12f);
      for (int c = 0; c < 64; ++c) {
        float4 a = ps4[c], d = pe4[c], p = pp4[c], gg = g4[c], bb = be4[c];
        bf16x4 o = { tobf((a.x + d.x + p.x - mu) * rstd * gg.x + bb.x),
                     tobf((a.y + d.y + p.y - mu) * rstd * gg.y + bb.y),
                     tobf((a.z + d.z + p.z - mu) * rstd * gg.z + bb.z),
                     tobf((a.w + d.w + p.w - mu) * rstd * gg.w + bb.w) };
        *(bf16x4*)&patch[rr][c * 4] = o;
      }
    }
  }

  int p = wave * 16 + lr;        // position handled by this lane for A frags
  int oi = p >> 3, oj = p & 7;

  // 2. 5x5 conv as implicit GEMM, full N=256 in registers (16 accums).
  // Slab s (0..199): dd = s>>3 (tap), k0 = (s&7)*32. Double-buffered async
  // prefetch of the [256][32] bf16 weight slab (8 x b128 per thread).
  bf16 (*WT0)[32] = (bf16(*)[32])stage;
  bf16 (*WT1)[32] = (bf16(*)[32])(stage + 8192);
  auto issue_slab = [&](int s, bf16 (*buf)[32]) {
    int dd = s >> 3, k0 = (s & 7) * 32;
    const bf16* wdd = convwT + (long)dd * 65536;
#pragma unroll
    for (int e = 0; e < 8; ++e) {
      int idx = tid + 128 * e;
      int row = idx >> 2, ch = (idx & 3) * 8;
      async_copy16(ldsoff(&buf[row][ch]), wdd + (long)row * 256 + k0 + ch);
    }
  };

  issue_slab(0, WT0);
  __syncthreads();               // patch complete (also covers LN stores)

  floatx8 acc[16] = {};
  for (int s = 0; s < 200; ++s) {
    bf16 (*cur)[32] = (s & 1) ? WT1 : WT0;
    if (s + 1 < 200) {
      issue_slab(s + 1, (s & 1) ? WT0 : WT1);
      wait_async_le8();          // slab s resident (async loads complete in order)
    } else {
      wait_async_0();
    }
    __syncthreads();
    int dd = s >> 3, k0 = (s & 7) * 32;
    int di = dd / 5, dj = dd % 5;
    FragU fa;
    const bf16* ap = &patch[(oi + di) * 12 + (oj + dj)][k0];
    fa.h[0] = *(const bf16x8*)(ap + 16 * lg);
    fa.h[1] = *(const bf16x8*)(ap + 16 * lg + 8);
#pragma unroll
    for (int q = 0; q < 16; ++q) {
      int nb = q >> 2, t = q & 3;
      FragU fb;
      const bf16* bp = &cur[nb * 64 + t * 16 + lr][16 * lg];
      fb.h[0] = *(const bf16x8*)(bp);
      fb.h[1] = *(const bf16x8*)(bp + 8);
      acc[q] = __builtin_amdgcn_wmma_f32_16x16x32_bf16(
          false, fa.v, false, fb.v, (short)0, acc[q], false, false);
    }
    __syncthreads();             // reads of cur done -> reusable at s+2
  }
#pragma unroll
  for (int q = 0; q < 16; ++q) {
    int nb = q >> 2, t = q & 3;
    int col = nb * 64 + t * 16 + lr;
    float bv = convb[col];
#pragma unroll
    for (int v = 0; v < 8; ++v) {
      int pr = wave * 16 + v + 8 * lg;
      Y[pr][col] = tobf(acc[q][v] + bv);
    }
  }
  __syncthreads();

  // 3. Score MLP + mask -> final S.
  score_mlp64(Y, (bf16(*)[D2C])stage, spart, W1T, b1, w2);
  if (tid < 64) {
    int i = i0 + (tid >> 3), j = j0 + (tid & 7);
    float S = spart[tid] + b2[0];
    float mm = masks[b * LL + i] * masks[b * LL + j] * tri[i * LL + j];
    Sout[((long)(b * LL + i)) * LL + j] = S - 10000.f * (1.f - mm);
  }
}

// ---------------------------------------------------------------------------
extern "C" void kernel_launch(void* const* d_in, const int* in_sizes, int n_in,
                              void* d_out, int out_size, void* d_ws, size_t ws_size,
                              hipStream_t stream)
{
  (void)in_sizes; (void)n_in; (void)out_size; (void)ws_size;
  const float* H_c      = (const float*)d_in[0];
  const float* masks    = (const float*)d_in[2];
  const float* be_e_b1  = (const float*)d_in[3];
  const float* be_e_b2  = (const float*)d_in[4];
  const float* be_e_w1  = (const float*)d_in[5];
  const float* be_e_w2  = (const float*)d_in[6];
  const float* be_s_b1  = (const float*)d_in[7];
  const float* be_s_b2  = (const float*)d_in[8];
  const float* be_s_w1  = (const float*)d_in[9];
  const float* be_s_w2  = (const float*)d_in[10];
  const float* conv_b   = (const float*)d_in[11];
  const float* conv_w   = (const float*)d_in[12];
  const float* se_e_be  = (const float*)d_in[13];
  const float* se_e_g   = (const float*)d_in[14];
  const float* se_e_sb  = (const float*)d_in[15];
  const float* se_e_sw  = (const float*)d_in[16];
  const float* se_s_be  = (const float*)d_in[17];
  const float* se_s_g   = (const float*)d_in[18];
  const float* se_s_sb  = (const float*)d_in[19];
  const float* se_s_sw  = (const float*)d_in[20];
  const float* se_x_be  = (const float*)d_in[21];
  const float* se_x_g   = (const float*)d_in[22];
  const float* se_x_sb  = (const float*)d_in[23];
  const float* se_x_sw  = (const float*)d_in[24];
  const float* sse_b1   = (const float*)d_in[29];
  const float* sse_b2   = (const float*)d_in[30];
  const float* sse_w1   = (const float*)d_in[31];
  const float* sse_w2   = (const float*)d_in[32];
  const float* sss_b1   = (const float*)d_in[37];
  const float* sss_b2   = (const float*)d_in[38];
  const float* sss_w1   = (const float*)d_in[39];
  const float* sss_w2   = (const float*)d_in[40];
  const float* ssx_b1   = (const float*)d_in[45];
  const float* ssx_b2   = (const float*)d_in[46];
  const float* ssx_w1   = (const float*)d_in[47];
  const float* ssx_w2   = (const float*)d_in[48];
  const float* w2e_b    = (const float*)d_in[49];
  const float* w2e_w    = (const float*)d_in[50];
  const float* w2s_b    = (const float*)d_in[51];
  const float* w2s_w    = (const float*)d_in[52];
  const float* pos_e    = (const float*)d_in[53];
  const float* pos_s    = (const float*)d_in[54];
  const float* pos_x    = (const float*)d_in[55];
  const float* tri      = (const float*)d_in[56];
  float* Sout = (float*)d_out;

  float* ws   = (float*)d_ws;
  float* tmp  = ws;                 // [768,768]
  float* hBs  = ws + 589824;        // [768,768]
  float* hBe  = ws + 1179648;       // [768,768]
  float* Bs2s = ws + 1769472;       // [768,256]
  float* Be2s = ws + 1966080;
  float* Bs2e = ws + 2162688;
  float* Be2e = ws + 2359296;
  float* Vs   = ws + 2555904;       // [768,768]
  float* Ve   = ws + 3145728;
  float* Gs   = ws + 3735552;       // [4,192,192]
  float* Ge   = ws + 3883008;       // [4,192,192] (stored transposed)
  float* BsA  = ws + 4030464;       // [768,768]
  float* BeA  = ws + 4620288;
  float* Bs2x = ws + 5210112;       // [768,256]
  float* Be2x = ws + 5406720;
  bf16* W1T_s  = (bf16*)(ws + 5603328);   // [256][256] bf16, transposed
  bf16* W1T_e  = W1T_s + 65536;
  bf16* W1T_x  = W1T_e + 65536;
  bf16* convwT = W1T_x + 65536;           // [25][256][256] bf16, [dd][co][ci]

  dim3 blk(128);
  auto gemm = [&](const float* A, const float* W, const float* bias, float* C,
                  int M, int N, int K, int act,
                  int batch = 1, long sA = 0, long sW = 0, long sC = 0) {
    dim3 grid(N / 64, M / 64, batch);
    gemm_bias_act<<<grid, blk, 0, stream>>>(A, W, bias, C, M, N, K, sA, sW, sC, act);
  };

  // Pre-convert score W1s and conv weights to transposed bf16 (L2-resident).
  w_to_bf16_t<<<dim3(256, 1),  dim3(256), 0, stream>>>(sss_w1, W1T_s);
  w_to_bf16_t<<<dim3(256, 1),  dim3(256), 0, stream>>>(sse_w1, W1T_e);
  w_to_bf16_t<<<dim3(256, 1),  dim3(256), 0, stream>>>(ssx_w1, W1T_x);
  w_to_bf16_t<<<dim3(256, 25), dim3(256), 0, stream>>>(conv_w, convwT);

  // Boundary MLPs: hBs = mlp(H_c, be_s), hBe = mlp(H_c, be_e)
  gemm(H_c, be_s_w1, be_s_b1, tmp, 768, 768, 768, 1);
  gemm(tmp, be_s_w2, be_s_b2, hBs, 768, 768, 768, 0);
  gemm(H_c, be_e_w1, be_e_b1, tmp, 768, 768, 768, 1);
  gemm(tmp, be_e_w2, be_e_b2, hBe, 768, 768, 768, 0);

  // Span-enum projections (both sides through the same sw, per reference quirk)
  gemm(hBs, se_s_sw, se_s_sb, Bs2s, 768, 256, 768, 0);
  gemm(hBe, se_s_sw, se_s_sb, Be2s, 768, 256, 768, 0);
  gemm(hBs, se_e_sw, se_e_sb, Bs2e, 768, 256, 768, 0);
  gemm(hBe, se_e_sw, se_e_sb, Be2e, 768, 256, 768, 0);

  // Value projections (both use hBs, per reference quirk)
  gemm(hBs, w2s_w, w2s_b, Vs, 768, 768, 768, 0);
  gemm(hBs, w2e_w, w2e_b, Ve, 768, 768, 768, 0);

  // Fused span-enum + score -> masked logits
  {
    dim3 grid(3, LL, 4);
    span_score_kernel<<<grid, blk, 0, stream>>>(
        Bs2s, Be2s, pos_s, se_s_g, se_s_be,
        W1T_s, sss_b1, sss_w2, sss_b2, masks, tri, Gs, 0);
    span_score_kernel<<<grid, blk, 0, stream>>>(
        Bs2e, Be2e, pos_e, se_e_g, se_e_be,
        W1T_e, sse_b1, sse_w2, sse_b2, masks, tri, Ge, 1);
  }

  // Row softmax (Ge stored transposed, so both are plain row softmax)
  softmax192<<<dim3(4 * LL), dim3(256), 0, stream>>>(Gs);
  softmax192<<<dim3(4 * LL), dim3(256), 0, stream>>>(Ge);

  // Attention aggregation: Bs = P_s @ Vs, Be = P_e @ Ve (batched)
  gemm(Gs, Vs, nullptr, BsA, 192, 768, 192, 0, 4,
       (long)192 * 192, (long)192 * 768, (long)192 * 768);
  gemm(Ge, Ve, nullptr, BeA, 192, 768, 192, 0, 4,
       (long)192 * 192, (long)192 * 768, (long)192 * 768);

  // se_x projections of the aggregated boundaries
  gemm(BsA, se_x_sw, se_x_sb, Bs2x, 768, 256, 768, 0);
  gemm(BeA, se_x_sw, se_x_sb, Be2x, 768, 256, 768, 0);

  // Fused span-enum + 5x5 conv + score -> final S
  {
    dim3 grid(LL / 8, LL / 8, 4);
    conv_score_kernel<<<grid, blk, 0, stream>>>(
        Bs2x, Be2x, pos_x, se_x_g, se_x_be, convwT, conv_b,
        W1T_x, ssx_b1, ssx_w2, ssx_b2, masks, tri, Sout);
  }
}
